// LSTM_77421080478321
// MI455X (gfx1250) — matrix operand order repaired and checked by reading, not compiled, not run on hardware
//
#include <hip/hip_runtime.h>
#include <hip/hip_bf16.h>
#include <math.h>

// ---------------------------------------------------------------------------
// Fused embedding + pos-enc + LSTM (WMMA bf16) + MLP head for gfx1250.
//
// B=128,T=1024,D=128,H=512,V=8,OUT=3.
// 8 workgroups x 256 threads (8 wave32). Workgroup g owns batch rows
// [16g,16g+16): private (h,c) state -> no inter-WG sync across the 1024
// sequential steps. Per step each wave computes 4 hidden-column tiles x 4
// gates with v_wmma_f32_16x16x32_bf16 (fused x@Wx K=128 + h@U K=512), does
// the LSTM cell update in registers (c never leaves VGPRs), and writes h
// (bf16) to a double-buffered LDS tile that is the next step's A-matrix.
// Weights (2.5MB bf16) are pre-swizzled into the exact 32x16 B-fragment
// lane layout so each fragment is one contiguous 32B load per lane,
// streamed from L2 every step. Position encodings are precomputed to a
// 512KB L2-resident table so the serial loop does no sin/cos/exp for x.
// ---------------------------------------------------------------------------

typedef __attribute__((ext_vector_type(16))) __bf16       v16bf;
typedef __attribute__((ext_vector_type(8)))  float        v8f;
typedef __attribute__((ext_vector_type(8)))  unsigned int v8u;
typedef __attribute__((ext_vector_type(4)))  unsigned int v4u;

#define B_    128
#define T_    1024
#define D_    128
#define H_    512
#define G4_   2048      // 4*H
#define NT_   128       // G4_/16 column tiles of z
#define ROWS_ 16        // batch rows per workgroup (WMMA M)
#define NWG_  (B_/ROWS_)

__device__ __forceinline__ unsigned int bf16_rne(float f) {
  unsigned int u = __float_as_uint(f);
  return (u + 0x7FFFu + ((u >> 16) & 1u)) >> 16;
}
__device__ __forceinline__ unsigned int pack2(float a, float b) {
  return bf16_rne(a) | (bf16_rne(b) << 16);
}
__device__ __forceinline__ float bf16_to_f32(unsigned short h) {
  return __uint_as_float(((unsigned int)h) << 16);
}
__device__ __forceinline__ float sigm(float x) { return 1.0f / (1.0f + __expf(-x)); }
__device__ __forceinline__ float tanh_fast(float x) { return 1.0f - 2.0f / (__expf(2.0f * x) + 1.0f); }

// ---------------------------------------------------------------------------
// Pre-swizzle fp32 row-major [K, 2048] weights into bf16 B-fragment layout.
// 32x16 bf16 B tile (ISA 7.12.2): lane l (N = l&15), VGPR v holds packed
// K pair {hi+2v,hi+2v+1} for v<4, {16+hi+2(v-4),..+1} for v>=4 (hi = 8 for lanes 16-31).
// dst flat: [((kt*NT + nt)*32 + lane)*8 + v]  -> 32B contiguous per lane.
// ---------------------------------------------------------------------------
__global__ void swizzle_B(const float* __restrict__ src, unsigned int* __restrict__ dst,
                          int ktiles, int ncols) {
  int id = blockIdx.x * blockDim.x + threadIdx.x;
  int ntiles = ncols >> 4;
  int total  = ktiles * ntiles * 256;
  if (id >= total) return;
  int v    = id & 7;
  int l    = (id >> 3) & 31;
  int tile = id >> 8;
  int nt   = tile % ntiles;
  int kt   = tile / ntiles;
  int col  = nt * 16 + (l & 15);
  int k0   = kt * 32 + ((l & 16) ? 8 : 0) + ((v < 4) ? 2 * v : 16 + 2 * (v - 4));
  dst[id] = pack2(src[k0 * ncols + col], src[(k0 + 1) * ncols + col]);
}

// Precompute transformer sinusoidal position encodings: pos[t*128 + c],
// c<64 -> sin(t*exp(-c*ln(1e4)/63)), c>=64 -> cos with (c-64).
__global__ void build_pos(float* __restrict__ pos) {
  int id = blockIdx.x * blockDim.x + threadIdx.x;
  if (id >= T_ * D_) return;
  int t = id >> 7, c = id & 127;
  const float log_inc = 0.14624447f;   // ln(1e4)/63
  pos[id] = (c < 64) ? __sinf((float)t * __expf(-(float)c * log_inc))
                     : __cosf((float)t * __expf(-(float)(c - 64) * log_inc));
}

// A fragment (16x32 bf16, ISA 7.12.2) from an LDS tile [16][stride] of bf16:
// lane l -> M = l&15; VGPR v<4: K = hi + 2v (hi = 8 for upper half-wave),
// v>=4: K = 16 + hi + 2(v-4). Even K -> two contiguous ds_load_b128.
__device__ __forceinline__ v16bf load_afrag(const unsigned short* lds, int stride,
                                            int kt, int lane) {
  int row = lane & 15;
  int hi  = (lane & 16) ? 8 : 0;
  const unsigned short* base = lds + row * stride + kt * 32;
  v8u a;
#pragma unroll
  for (int v = 0; v < 8; ++v) {
    int kl = (v < 4) ? (hi + 2 * v) : (16 + hi + 2 * (v - 4));
    a[v] = *(const unsigned int*)(base + kl);
  }
  return __builtin_bit_cast(v16bf, a);
}

__global__ __launch_bounds__(256) void lstm_fused(
    const int* __restrict__ X, const float* __restrict__ emb,
    const unsigned int* __restrict__ Wxs, const unsigned int* __restrict__ Us,
    const float* __restrict__ pos, const float* __restrict__ bias,
    const float* __restrict__ W2, const float* __restrict__ b2,
    const float* __restrict__ W3, const float* __restrict__ b3,
    const float* __restrict__ Wo, const float* __restrict__ bo,
    float* __restrict__ out)
{
  __shared__ unsigned short h_lds[2][ROWS_ * H_];   // bf16 h, double buffered
  __shared__ unsigned short x_lds[ROWS_ * D_];      // bf16 (emb + posenc) tile
  __shared__ float o1[ROWS_ * 256];
  __shared__ float o2[ROWS_ * 128];

  const int tid  = threadIdx.x;
  const int lane = tid & 31;
  const int wave = tid >> 5;
  const int wg   = blockIdx.x;            // batch rows [wg*16, wg*16+16)

  for (int i = tid; i < ROWS_ * H_; i += 256) h_lds[0][i] = 0;

  float cst[4][8];                         // c state: 4 hidden tiles x 8 rows
#pragma unroll
  for (int n = 0; n < 4; ++n)
#pragma unroll
    for (int r = 0; r < 8; ++r) cst[n][r] = 0.0f;

  float bval[4][4];                        // bias per (hidden tile, gate)
#pragma unroll
  for (int nl = 0; nl < 4; ++nl) {
    int nh = wave * 4 + nl;
#pragma unroll
    for (int g = 0; g < 4; ++g)
      bval[nl][g] = bias[g * H_ + nh * 16 + (lane & 15)];
  }

  // x-tile build mapping: thread owns row = tid/16, 8 consecutive cols.
  const int xrow = tid >> 4;
  const int xc0  = (tid & 15) * 8;
  const int* Xrow = X + (wg * ROWS_ + xrow) * T_;
  __syncthreads();

  int cur = 0;
  for (int t = 0; t < T_; ++t) {
    // x_t tile: emb[X[b,t]] + pos[t]  ->  bf16 LDS. One token load, two
    // float4 emb + two float4 pos loads, one 16B LDS store per thread.
    {
      int tok = Xrow[t];
      const float4* ep = (const float4*)(emb + tok * D_ + xc0);
      const float4* pp = (const float4*)(pos + t * D_ + xc0);
      float4 e0 = ep[0], e1 = ep[1];
      float4 p0 = pp[0], p1 = pp[1];
      v4u xv;
      xv[0] = pack2(e0.x + p0.x, e0.y + p0.y);
      xv[1] = pack2(e0.z + p0.z, e0.w + p0.w);
      xv[2] = pack2(e1.x + p1.x, e1.y + p1.y);
      xv[3] = pack2(e1.z + p1.z, e1.w + p1.w);
      *(v4u*)(x_lds + xrow * D_ + xc0) = xv;
    }
    __syncthreads();

#pragma unroll 1
    for (int nl = 0; nl < 4; ++nl) {
      int nh = wave * 4 + nl;              // global hidden tile 0..31
      v8f acc[4];
#pragma unroll
      for (int g = 0; g < 4; ++g) {
        float bv = bval[nl][g];
#pragma unroll
        for (int r = 0; r < 8; ++r) acc[g][r] = bv;
      }
      // z += x_t @ Wx   (K = 128)
#pragma unroll 1
      for (int kt = 0; kt < 4; ++kt) {
        v16bf af = load_afrag(x_lds, D_, kt, lane);
#pragma unroll
        for (int g = 0; g < 4; ++g) {
          v8u bu = *(const v8u*)(Wxs + ((kt * NT_ + (g * 32 + nh)) * 32 + lane) * 8);
          v16bf bfr = __builtin_bit_cast(v16bf, bu);
          acc[g] = __builtin_amdgcn_wmma_f32_16x16x32_bf16(
              false, af, false, bfr, (short)0, acc[g], false, false);
        }
      }
      // z += h_{t-1} @ U   (K = 512); unroll 2 for deeper load/WMMA overlap
#pragma unroll 2
      for (int kt = 0; kt < 16; ++kt) {
        v16bf af = load_afrag(h_lds[cur], H_, kt, lane);
#pragma unroll
        for (int g = 0; g < 4; ++g) {
          v8u bu = *(const v8u*)(Us + ((kt * NT_ + (g * 32 + nh)) * 32 + lane) * 8);
          v16bf bfr = __builtin_bit_cast(v16bf, bu);
          acc[g] = __builtin_amdgcn_wmma_f32_16x16x32_bf16(
              false, af, false, bfr, (short)0, acc[g], false, false);
        }
      }
      // LSTM cell update in registers. C/D layout: lane = N (col), VGPR r ->
      // row r (lanes 0-15) or r+8 (lanes 16-31). Gate order i,f,g,o.
      int colg = nh * 16 + (lane & 15);
      int rowb = (lane & 16) ? 8 : 0;
#pragma unroll
      for (int r = 0; r < 8; ++r) {
        float c = sigm(acc[1][r]) * cst[nl][r] + sigm(acc[0][r]) * tanh_fast(acc[2][r]);
        cst[nl][r] = c;
        float h = sigm(acc[3][r]) * tanh_fast(c);
        h_lds[cur ^ 1][(rowb + r) * H_ + colg] = (unsigned short)bf16_rne(h);
      }
    }
    __syncthreads();
    cur ^= 1;
  }

  // ---- MLP head on this workgroup's 16 rows (weights L2-resident). ----
  // h_last read straight from the final bf16 LDS buffer (runs once; VALU
  // cost negligible vs the 1024-step WMMA recurrence).
  const unsigned short* hfin = h_lds[cur];
  for (int j = tid; j < ROWS_ * 256; j += 256) {
    int row = j >> 8, col = j & 255;
    float s = b2[col];
    for (int k = 0; k < H_; ++k) s += bf16_to_f32(hfin[row * H_ + k]) * W2[k * 256 + col];
    o1[j] = fmaxf(s, 0.0f);
  }
  __syncthreads();
  for (int j = tid; j < ROWS_ * 128; j += 256) {
    int row = j >> 7, col = j & 127;
    float s = b3[col];
    for (int k = 0; k < 256; ++k) s += o1[row * 256 + k] * W3[k * 128 + col];
    o2[j] = fmaxf(s, 0.0f);
  }
  __syncthreads();
  for (int j = tid; j < ROWS_ * 3; j += 256) {
    int row = j / 3, col = j % 3;
    float s = bo[col];
    for (int k = 0; k < 128; ++k) s += o2[row * 128 + k] * Wo[k * 3 + col];
    out[(wg * ROWS_ + row) * 3 + col] = s;
  }
}

extern "C" void kernel_launch(void* const* d_in, const int* in_sizes, int n_in,
                              void* d_out, int out_size, void* d_ws, size_t ws_size,
                              hipStream_t stream) {
  const int*   X   = (const int*)  d_in[0];
  const float* emb = (const float*)d_in[1];
  const float* Wx  = (const float*)d_in[2];
  const float* U   = (const float*)d_in[3];
  const float* b   = (const float*)d_in[4];
  const float* W2  = (const float*)d_in[5];
  const float* b2  = (const float*)d_in[6];
  const float* W3  = (const float*)d_in[7];
  const float* b3  = (const float*)d_in[8];
  const float* Wo  = (const float*)d_in[9];
  const float* bo  = (const float*)d_in[10];
  float* out = (float*)d_out;

  // Workspace: pre-swizzled bf16 weight fragments + fp32 pos-enc table.
  // U:   16 k-tiles * 128 n-tiles * 256 u32 = 2 MB
  // Wx:   4 k-tiles * 128 n-tiles * 256 u32 = 512 KB
  // pos: 1024 * 128 floats                  = 512 KB
  unsigned int* Us  = (unsigned int*)d_ws;
  unsigned int* Wxs = Us + 16 * NT_ * 256;
  float*        pos = (float*)(Wxs + 4 * NT_ * 256);

  int totU = 16 * NT_ * 256;
  int totW = 4 * NT_ * 256;
  int totP = T_ * D_;
  swizzle_B<<<(totU + 255) / 256, 256, 0, stream>>>(U,  Us,  16, G4_);
  swizzle_B<<<(totW + 255) / 256, 256, 0, stream>>>(Wx, Wxs, 4,  G4_);
  build_pos<<<(totP + 255) / 256, 256, 0, stream>>>(pos);

  lstm_fused<<<NWG_, 256, 0, stream>>>(X, emb, Wxs, Us, pos, b,
                                       W2, b2, W3, b3, Wo, bo, out);
}
